// MemoryDecoder_70978629534057
// MI455X (gfx1250) — matrix operand order repaired and checked by reading, not compiled, not run on hardware
//
#include <hip/hip_runtime.h>

#define HH 64
#define WW 96
#define HWPIX 6144

typedef __attribute__((ext_vector_type(16))) _Float16 v16h;
typedef __attribute__((ext_vector_type(8)))  float    v8f;

__device__ __forceinline__ float act_f(float x, int act) {
  if (act == 1) return fmaxf(x, 0.0f);
  if (act == 2) return 0.5f * x * (1.0f + erff(x * 0.70710678118654752f));
  if (act == 3) return tanhf(x);
  if (act == 4) return 1.0f / (1.0f + __expf(-x));
  return x;
}

// A-fragment load (16x32 f16): lane holds K = half*16..+15 contiguous
__device__ __forceinline__ v16h ldA(const _Float16* __restrict__ p) {
  v16h a;
#pragma unroll
  for (int j = 0; j < 8; ++j) { a[j] = p[j]; a[8 + j] = p[8 + j]; }
  return a;
}
// B-fragment load (32x16 f16, same k order as ldA): lane holds K = half*16..+15 contiguous
__device__ __forceinline__ v16h ldB(const _Float16* __restrict__ p) {
  v16h b;
#pragma unroll
  for (int j = 0; j < 16; ++j) b[j] = p[j];
  return b;
}

// ---------------------------------------------------------------------------
// WMMA GEMM: out[M,Nout] = act((A[M,Kpad] * W[Npad,Kpad]^T + bias) * scale) + resid
// block = 128 threads (4 waves) covering 128x64; each wave: 32x64 (8 accum tiles).
// Software-pipelined K loop (register double buffering).
// out2 (f16) optional; tr2!=0 stores it transposed: out2[col*ld2 + row].
// ---------------------------------------------------------------------------
__global__ void k_gemm(const _Float16* __restrict__ A,
                       const _Float16* __restrict__ Bw,
                       float* __restrict__ out,
                       _Float16* __restrict__ out2, int ld2, int tr2,
                       const float* __restrict__ bias,
                       const float* __restrict__ resid,
                       int M, int Nout, int Kpad, int act, float scale)
{
  const int wid  = threadIdx.x >> 5;
  const int lane = threadIdx.x & 31;
  const int half = lane >> 4;
  const int mn   = lane & 15;
  const int mbase = blockIdx.x * 128 + wid * 32;
  const int nbase = blockIdx.y * 64;
  if (mbase >= M) return;
  v8f acc[8] = {};
  const _Float16* ar0 = A + (size_t)(mbase + mn) * Kpad + half * 16;
  const _Float16* ar1 = ar0 + (size_t)16 * Kpad;
  const _Float16* br0 = Bw + (size_t)(nbase + mn) * Kpad + half * 16;
  const size_t bstep = (size_t)16 * Kpad;

  v16h a0 = ldA(ar0), a1 = ldA(ar1);
  v16h b0 = ldB(br0), b1 = ldB(br0 + bstep), b2 = ldB(br0 + 2 * bstep), b3 = ldB(br0 + 3 * bstep);
  int k0 = 0;
  while (true) {
    const int kn = k0 + 32;
    v16h na0, na1, nb0, nb1, nb2, nb3;
    if (kn < Kpad) {  // prefetch next K-step while current WMMAs run
      na0 = ldA(ar0 + kn);             na1 = ldA(ar1 + kn);
      nb0 = ldB(br0 + kn);             nb1 = ldB(br0 + bstep + kn);
      nb2 = ldB(br0 + 2 * bstep + kn); nb3 = ldB(br0 + 3 * bstep + kn);
    }
    acc[0] = __builtin_amdgcn_wmma_f32_16x16x32_f16(false, a0, false, b0, (short)0, acc[0], false, false);
    acc[1] = __builtin_amdgcn_wmma_f32_16x16x32_f16(false, a0, false, b1, (short)0, acc[1], false, false);
    acc[2] = __builtin_amdgcn_wmma_f32_16x16x32_f16(false, a0, false, b2, (short)0, acc[2], false, false);
    acc[3] = __builtin_amdgcn_wmma_f32_16x16x32_f16(false, a0, false, b3, (short)0, acc[3], false, false);
    acc[4] = __builtin_amdgcn_wmma_f32_16x16x32_f16(false, a1, false, b0, (short)0, acc[4], false, false);
    acc[5] = __builtin_amdgcn_wmma_f32_16x16x32_f16(false, a1, false, b1, (short)0, acc[5], false, false);
    acc[6] = __builtin_amdgcn_wmma_f32_16x16x32_f16(false, a1, false, b2, (short)0, acc[6], false, false);
    acc[7] = __builtin_amdgcn_wmma_f32_16x16x32_f16(false, a1, false, b3, (short)0, acc[7], false, false);
    if (kn >= Kpad) break;
    a0 = na0; a1 = na1; b0 = nb0; b1 = nb1; b2 = nb2; b3 = nb3;
    k0 = kn;
  }
#pragma unroll
  for (int i = 0; i < 2; ++i)
#pragma unroll
    for (int t = 0; t < 4; ++t) {
      int col = nbase + t * 16 + mn;
      if (col >= Nout) continue;
      float bv = bias ? bias[col] : 0.0f;
      v8f av = acc[i * 4 + t];
#pragma unroll
      for (int r = 0; r < 8; ++r) {
        int row = mbase + i * 16 + r + 8 * half;
        float v = (av[r] + bv) * scale;
        v = act_f(v, act);
        if (resid) v += resid[(size_t)row * Nout + col];
        out[(size_t)row * Nout + col] = v;
        if (out2) {
          if (tr2) out2[(size_t)col * ld2 + row] = (_Float16)v;
          else     out2[(size_t)row * ld2 + col] = (_Float16)v;
        }
      }
    }
}

// ---------------------------------------------------------------------------
// Fused GMA attention-aggregate (flash style): O = softmax(Q K^T) V
// Q,K f16 [N,128]; Vt f16 TRANSPOSED [128,N] so PV B-tiles are contiguous.
// One wave per 16-row query tile; 32 keys per step; V loads issued before the
// softmax VALU work so their latency is hidden.
// ---------------------------------------------------------------------------
__global__ void k_flash(const _Float16* __restrict__ Q,
                        const _Float16* __restrict__ Kt,
                        const _Float16* __restrict__ Vt,
                        float* __restrict__ O, int N)
{
  __shared__ _Float16 psh[4][512];
  const int wid  = threadIdx.x >> 5;
  const int lane = threadIdx.x & 31;
  const int half = lane >> 4;
  const int mn   = lane & 15;
  const int mbase = (blockIdx.x * 4 + wid) * 16;
  v16h qa[4];
#pragma unroll
  for (int s = 0; s < 4; ++s)
    qa[s] = ldA(Q + (size_t)(mbase + mn) * 128 + s * 32 + half * 16);
  float rmax[8], rsum[8], fac[8];
#pragma unroll
  for (int r = 0; r < 8; ++r) { rmax[r] = -1e30f; rsum[r] = 0.0f; }
  v8f o[8] = {};
  for (int n0 = 0; n0 < N; n0 += 32) {
    // scores for 32 keys (two 16x16 C tiles)
    v8f s0 = {}, s1 = {};
#pragma unroll
    for (int s = 0; s < 4; ++s) {
      v16h b0 = ldB(Kt + (size_t)(n0 + mn) * 128 + s * 32 + half * 16);
      v16h b1 = ldB(Kt + (size_t)(n0 + 16 + mn) * 128 + s * 32 + half * 16);
      s0 = __builtin_amdgcn_wmma_f32_16x16x32_f16(false, qa[s], false, b0, (short)0, s0, false, false);
      s1 = __builtin_amdgcn_wmma_f32_16x16x32_f16(false, qa[s], false, b1, (short)0, s1, false, false);
    }
    // issue V-tile loads now (contiguous thanks to transpose); latency overlaps softmax
    v16h bv[8];
#pragma unroll
    for (int t = 0; t < 8; ++t)
      bv[t] = ldB(Vt + (size_t)(t * 16 + mn) * N + n0 + half * 16);
    // online softmax; row (r + 8*half) lives across the 16 lanes of this half
#pragma unroll
    for (int r = 0; r < 8; ++r) {
      float c = fmaxf(s0[r], s1[r]);
#pragma unroll
      for (int off = 8; off > 0; off >>= 1) c = fmaxf(c, __shfl_xor(c, off, 16));
      float mnew = fmaxf(rmax[r], c);
      fac[r] = __expf(rmax[r] - mnew);
      rmax[r] = mnew;
      float p0 = __expf(s0[r] - mnew);
      float p1 = __expf(s1[r] - mnew);
      s0[r] = p0; s1[r] = p1;
      float ps = p0 + p1;
#pragma unroll
      for (int off = 8; off > 0; off >>= 1) ps += __shfl_xor(ps, off, 16);
      rsum[r] = rsum[r] * fac[r] + ps;
    }
#pragma unroll
    for (int t = 0; t < 8; ++t)
#pragma unroll
      for (int r = 0; r < 8; ++r) o[t][r] *= fac[r];
    // P: C-layout -> A-layout through per-wave LDS
#pragma unroll
    for (int r = 0; r < 8; ++r) {
      psh[wid][(r + 8 * half) * 32 + mn]      = (_Float16)s0[r];
      psh[wid][(r + 8 * half) * 32 + 16 + mn] = (_Float16)s1[r];
    }
    __syncthreads();
    v16h pa;
#pragma unroll
    for (int j = 0; j < 8; ++j) {
      pa[j]     = psh[wid][mn * 32 + half * 16 + j];
      pa[8 + j] = psh[wid][mn * 32 + half * 16 + 8 + j];
    }
    __syncthreads();
#pragma unroll
    for (int t = 0; t < 8; ++t)
      o[t] = __builtin_amdgcn_wmma_f32_16x16x32_f16(false, pa, false, bv[t], (short)0, o[t], false, false);
  }
#pragma unroll
  for (int t = 0; t < 8; ++t)
#pragma unroll
    for (int r = 0; r < 8; ++r)
      O[(size_t)(mbase + r + 8 * half) * 128 + t * 16 + mn] = o[t][r] / rsum[r];
}

// ---------------------------------------------------------------------------
// Weight reorder OIHW -> f16 [Npad,Kpad], K index = (ky*KW+kx)*I + i, zero padded.
// Also used (KH=KW=1) for plain [M,C] f32->f16 conversion.
// ---------------------------------------------------------------------------
__global__ void k_wconv(const float* __restrict__ w, _Float16* __restrict__ out,
                        int O, int I, int KH, int KW, int Npad, int Kpad)
{
  long long idx = (long long)blockIdx.x * blockDim.x + threadIdx.x;
  if (idx >= (long long)Npad * Kpad) return;
  int n = (int)(idx / Kpad), k = (int)(idx - (long long)n * Kpad);
  int K = KH * KW * I;
  float val = 0.0f;
  if (n < O && k < K) {
    int pos = k / I, c = k - pos * I;
    int ky = pos / KW, kx = pos - ky * KW;
    val = w[(((size_t)n * I + c) * KH + ky) * KW + kx];
  }
  out[idx] = (_Float16)val;
}

__global__ void k_nchw2hwc(const float* __restrict__ in, _Float16* __restrict__ out, int C)
{
  long long idx = (long long)blockIdx.x * blockDim.x + threadIdx.x;
  if (idx >= (long long)HWPIX * C) return;
  int m = (int)(idx / C), c = (int)(idx - (long long)m * C);
  out[(size_t)m * C + c] = (_Float16)in[(size_t)c * HWPIX + m];
}

// im2col on up to 3 channel-concatenated HWC f32 tensors -> f16 [HW, Kpad]
__global__ void k_im2col3(const float* __restrict__ p0, int C0,
                          const float* __restrict__ p1, int C1,
                          const float* __restrict__ p2, int C2,
                          _Float16* __restrict__ out, int Kpad,
                          int KH, int KW, int padH, int padW)
{
  long long idx = (long long)blockIdx.x * blockDim.x + threadIdx.x;
  if (idx >= (long long)HWPIX * Kpad) return;
  int m = (int)(idx / Kpad), k = (int)(idx - (long long)m * Kpad);
  int Ct = C0 + C1 + C2;
  int K = KH * KW * Ct;
  float val = 0.0f;
  if (k < K) {
    int pos = k / Ct, c = k - pos * Ct;
    int ky = pos / KW, kx = pos - ky * KW;
    int y = m / WW + ky - padH;
    int x = m % WW + kx - padW;
    if (y >= 0 && y < HH && x >= 0 && x < WW) {
      const float* src; int cc, Cs;
      if (c < C0)           { src = p0; cc = c;           Cs = C0; }
      else if (c < C0 + C1) { src = p1; cc = c - C0;      Cs = C1; }
      else                  { src = p2; cc = c - C0 - C1; Cs = C2; }
      val = src[(size_t)(y * WW + x) * Cs + cc];
    }
  }
  out[(size_t)m * Kpad + k] = (_Float16)val;
}

// channel concat (up to 3) -> f32 and/or f16
__global__ void k_pack3(const float* __restrict__ p0, int C0,
                        const float* __restrict__ p1, int C1,
                        const float* __restrict__ p2, int C2,
                        float* __restrict__ out, _Float16* __restrict__ outh, int Cpad)
{
  long long idx = (long long)blockIdx.x * blockDim.x + threadIdx.x;
  if (idx >= (long long)HWPIX * Cpad) return;
  int m = (int)(idx / Cpad), c = (int)(idx - (long long)m * Cpad);
  float v = 0.0f;
  if (c < C0)                     v = p0[(size_t)m * C0 + c];
  else if (c < C0 + C1)           v = p1[(size_t)m * C1 + (c - C0)];
  else if (c < C0 + C1 + C2)      v = p2[(size_t)m * C2 + (c - C0 - C1)];
  if (out)  out[(size_t)m * Cpad + c] = v;
  if (outh) outh[(size_t)m * Cpad + c] = (_Float16)v;
}

__global__ void k_split_ctx(const float* __restrict__ proj, float* __restrict__ net,
                            float* __restrict__ inp, _Float16* __restrict__ inph)
{
  int idx = blockIdx.x * blockDim.x + threadIdx.x;
  if (idx >= HWPIX * 128) return;
  int m = idx >> 7, c = idx & 127;
  float nv = tanhf(proj[(size_t)m * 256 + c]);
  float iv = fmaxf(proj[(size_t)m * 256 + 128 + c], 0.0f);
  net[idx] = nv;
  inp[idx] = iv;
  inph[idx] = (_Float16)iv;
}

__global__ void k_split_qk(const float* __restrict__ qk, _Float16* __restrict__ qh,
                           _Float16* __restrict__ kh)
{
  int idx = blockIdx.x * blockDim.x + threadIdx.x;
  if (idx >= HWPIX * 128) return;
  int m = idx >> 7, c = idx & 127;
  qh[idx] = (_Float16)(qk[(size_t)m * 256 + c] * 0.08838834764831845f); // 128^-0.5
  kh[idx] = (_Float16)(qk[(size_t)m * 256 + 128 + c]);
}

// bilinear 9x9 cost-volume sampling; channel ch = i*9+j -> offset (d[i], d[j])
__global__ void k_sample(const float* __restrict__ cmap, const float* __restrict__ coords,
                         float* __restrict__ cf32, _Float16* __restrict__ cf16)
{
  int idx = blockIdx.x * blockDim.x + threadIdx.x;
  if (idx >= HWPIX * 96) return;
  int m = idx / 96, ch = idx - (idx / 96) * 96;
  if (ch >= 81) { cf16[(size_t)m * 96 + ch] = (_Float16)0.0f; return; }
  float xs = coords[2 * m]     + (float)(ch / 9) - 4.0f;
  float ys = coords[2 * m + 1] + (float)(ch % 9) - 4.0f;
  float x0 = floorf(xs), y0 = floorf(ys);
  float fx = xs - x0, fy = ys - y0;
  const float* img = cmap + (size_t)m * (HH * WW);
  float acc = 0.0f;
#pragma unroll
  for (int dy = 0; dy < 2; ++dy)
#pragma unroll
    for (int dx = 0; dx < 2; ++dx) {
      float xi = x0 + dx, yi = y0 + dy;
      float wgt = (dx ? fx : 1.0f - fx) * (dy ? fy : 1.0f - fy);
      if (xi >= 0.0f && xi <= (float)(WW - 1) && yi >= 0.0f && yi <= (float)(HH - 1))
        acc += wgt * img[(int)yi * WW + (int)xi];
    }
  cf32[(size_t)m * 81 + ch] = acc;
  cf16[(size_t)m * 96 + ch] = (_Float16)acc;
}

// LayerNorm(64) + sine positional embedding (added), f16 out
__global__ void k_qln_pos(const float* __restrict__ q, const float* __restrict__ coords,
                          const float* __restrict__ g, const float* __restrict__ b,
                          _Float16* __restrict__ out)
{
  int m = blockIdx.x * blockDim.x + threadIdx.x;
  if (m >= HWPIX) return;
  const float* x = q + (size_t)m * 64;
  float mean = 0.0f;
  for (int c = 0; c < 64; ++c) mean += x[c];
  mean *= (1.0f / 64.0f);
  float var = 0.0f;
  for (int c = 0; c < 64; ++c) { float d = x[c] - mean; var += d * d; }
  var *= (1.0f / 64.0f);
  float rstd = rsqrtf(var + 1e-5f);
  float cx = coords[2 * m], cy = coords[2 * m + 1];
  const float s = 3.14f / 200.0f;
  for (int c = 0; c < 64; ++c) {
    float ln = (x[c] - mean) * rstd * g[c] + b[c];
    int k = c & 15;
    float e;
    if (c < 16)      e = __sinf(cx * (float)k * s);
    else if (c < 32) e = __cosf(cx * (float)k * s);
    else if (c < 48) e = __sinf(cy * (float)k * s);
    else             e = __cosf(cy * (float)k * s);
    out[(size_t)m * 64 + c] = (_Float16)(ln + e);
  }
}

__global__ void k_ln64(const float* __restrict__ xin, const float* __restrict__ g,
                       const float* __restrict__ b, _Float16* __restrict__ out)
{
  int m = blockIdx.x * blockDim.x + threadIdx.x;
  if (m >= HWPIX) return;
  const float* x = xin + (size_t)m * 64;
  float mean = 0.0f;
  for (int c = 0; c < 64; ++c) mean += x[c];
  mean *= (1.0f / 64.0f);
  float var = 0.0f;
  for (int c = 0; c < 64; ++c) { float d = x[c] - mean; var += d * d; }
  var *= (1.0f / 64.0f);
  float rstd = rsqrtf(var + 1e-5f);
  for (int c = 0; c < 64; ++c)
    out[(size_t)m * 64 + c] = (_Float16)((x[c] - mean) * rstd * g[c] + b[c]);
}

// per-pixel 8-head / 8-key / 8-dim cross attention
__global__ void k_sattn(const float* __restrict__ qp, const float* __restrict__ kmem,
                        const float* __restrict__ vmem, float* __restrict__ xo)
{
  int idx = blockIdx.x * blockDim.x + threadIdx.x;
  if (idx >= HWPIX * 8) return;
  int m = idx >> 3, hd = idx & 7;
  const float* kb = kmem + (size_t)m * 8 * 64 + hd * 8;
  const float* vb = vmem + (size_t)m * 8 * 64 + hd * 8;
  float qv[8];
#pragma unroll
  for (int d = 0; d < 8; ++d) qv[d] = qp[(size_t)m * 64 + hd * 8 + d];
  float sc[8], mx = -1e30f;
#pragma unroll
  for (int t = 0; t < 8; ++t) {
    float s = 0.0f;
#pragma unroll
    for (int d = 0; d < 8; ++d) s += qv[d] * kb[t * 64 + d];
    s *= 0.3535533905932738f; // 8^-0.5
    sc[t] = s; mx = fmaxf(mx, s);
  }
  float den = 0.0f;
#pragma unroll
  for (int t = 0; t < 8; ++t) { sc[t] = __expf(sc[t] - mx); den += sc[t]; }
  float rd = 1.0f / den;
#pragma unroll
  for (int d = 0; d < 8; ++d) {
    float o = 0.0f;
#pragma unroll
    for (int t = 0; t < 8; ++t) o += sc[t] * vb[t * 64 + d];
    xo[(size_t)m * 64 + hd * 8 + d] = o * rd;
  }
}

__global__ void k_rh(const float* __restrict__ r, const float* __restrict__ net,
                     float* __restrict__ rh)
{
  int idx = blockIdx.x * blockDim.x + threadIdx.x;
  if (idx >= HWPIX * 128) return;
  rh[idx] = r[idx] * net[idx];
}

__global__ void k_gruout(float* __restrict__ net, const float* __restrict__ zb,
                         const float* __restrict__ qb)
{
  int idx = blockIdx.x * blockDim.x + threadIdx.x;
  if (idx >= HWPIX * 128) return;
  float z = zb[idx];
  net[idx] = (1.0f - z) * net[idx] + z * qb[idx];
}

__global__ void k_mfg(const float* __restrict__ mf, const float* __restrict__ agg,
                      const float* __restrict__ gamma, float* __restrict__ mfg)
{
  int idx = blockIdx.x * blockDim.x + threadIdx.x;
  if (idx >= HWPIX * 128) return;
  mfg[idx] = mf[idx] + gamma[0] * agg[idx];
}

__global__ void k_cinit(float* __restrict__ coords)
{
  int m = blockIdx.x * blockDim.x + threadIdx.x;
  if (m >= HWPIX) return;
  coords[2 * m]     = (float)(m % WW);
  coords[2 * m + 1] = (float)(m / WW);
}

__global__ void k_flowk(const float* __restrict__ coords, float* __restrict__ flow)
{
  int m = blockIdx.x * blockDim.x + threadIdx.x;
  if (m >= HWPIX) return;
  flow[2 * m]     = coords[2 * m]     - (float)(m % WW);
  flow[2 * m + 1] = coords[2 * m + 1] - (float)(m / WW);
}

__global__ void k_cadd(float* __restrict__ coords, const float* __restrict__ dflow)
{
  int idx = blockIdx.x * blockDim.x + threadIdx.x;
  if (idx >= HWPIX * 2) return;
  coords[idx] += dflow[idx];
}

// convex upsampling: out (2,512,768)
__global__ void k_upsample(const float* __restrict__ flow, const float* __restrict__ upmask,
                           float* __restrict__ out)
{
  int idx = blockIdx.x * blockDim.x + threadIdx.x;
  if (idx >= HWPIX * 64) return;
  int m = idx >> 6, pq = idx & 63;
  int h = m / WW, w = m % WW;
  float e[9], mx = -1e30f;
#pragma unroll
  for (int k = 0; k < 9; ++k) {
    e[k] = upmask[(size_t)m * 576 + k * 64 + pq];
    mx = fmaxf(mx, e[k]);
  }
  float den = 0.0f;
#pragma unroll
  for (int k = 0; k < 9; ++k) { e[k] = __expf(e[k] - mx); den += e[k]; }
  float rd = 1.0f / den;
  float a0 = 0.0f, a1 = 0.0f;
#pragma unroll
  for (int k = 0; k < 9; ++k) {
    int yy = h + k / 3 - 1, xx = w + k % 3 - 1;
    float v0 = 0.0f, v1 = 0.0f;
    if (yy >= 0 && yy < HH && xx >= 0 && xx < WW) {
      v0 = 8.0f * flow[(size_t)(yy * WW + xx) * 2];
      v1 = 8.0f * flow[(size_t)(yy * WW + xx) * 2 + 1];
    }
    float p = e[k] * rd;
    a0 += p * v0; a1 += p * v1;
  }
  int pp = pq >> 3, qq = pq & 7;
  int oy = h * 8 + pp, ox = w * 8 + qq;
  out[(size_t)oy * 768 + ox] = a0;
  out[(size_t)(512 * 768) + (size_t)oy * 768 + ox] = a1;
}

__global__ void k_outflow(const float* __restrict__ flow, float* __restrict__ out)
{
  int idx = blockIdx.x * blockDim.x + threadIdx.x;
  if (idx >= 2 * HWPIX) return;
  int c = idx / HWPIX, m = idx - c * HWPIX;
  out[idx] = flow[(size_t)m * 2 + c];
}

// ---------------------------------------------------------------------------
// Host orchestration
// ---------------------------------------------------------------------------
enum {
  IN_COST_MEMORY = 0, IN_CONTEXT, IN_COST_MAPS,
  P_PROJ_W, P_PROJ_B, P_FTE1_W, P_FTE1_B, P_FTE2_W, P_FTE2_B,
  P_CA_N1_G, P_CA_N1_B, P_CA_N2_G, P_CA_N2_B,
  P_CA_Q_W, P_CA_Q_B, P_CA_K_W, P_CA_K_B, P_CA_V_W, P_CA_V_B,
  P_CA_P_W, P_CA_P_B, P_CA_F1_W, P_CA_F1_B, P_CA_F2_W, P_CA_F2_B,
  P_ATT_QK_W,
  P_ENC_C1_W, P_ENC_C1_B, P_ENC_C2_W, P_ENC_C2_B,
  P_ENC_F1_W, P_ENC_F1_B, P_ENC_F2_W, P_ENC_F2_B,
  P_ENC_O_W, P_ENC_O_B, P_AGG_V_W, P_AGG_GAMMA,
  P_GRU_Z1_W, P_GRU_Z1_B, P_GRU_R1_W, P_GRU_R1_B, P_GRU_Q1_W, P_GRU_Q1_B,
  P_GRU_Z2_W, P_GRU_Z2_B, P_GRU_R2_W, P_GRU_R2_B, P_GRU_Q2_W, P_GRU_Q2_B,
  P_FH1_W, P_FH1_B, P_FH2_W, P_FH2_B,
  P_MASK1_W, P_MASK1_B, P_MASK2_W, P_MASK2_B,
  IN_QDIM
};

static inline unsigned g256(long long n) { return (unsigned)((n + 255) / 256); }

extern "C" void kernel_launch(void* const* d_in, const int* in_sizes, int n_in,
                              void* d_out, int out_size, void* d_ws, size_t ws_size,
                              hipStream_t stream)
{
  (void)in_sizes; (void)n_in; (void)out_size; (void)ws_size;
  auto P = [&](int i) { return (const float*)d_in[i]; };

  char* base = (char*)d_ws;
  size_t cur = 0;
  auto allocB = [&](size_t bytes) -> void* {
    void* p = base + cur;
    cur = (cur + bytes + 255) & ~(size_t)255;
    return p;
  };
  auto allocF = [&](size_t n) { return (float*)allocB(n * 4); };
  auto allocH = [&](size_t n) { return (_Float16*)allocB(n * 2); };

  auto wconv = [&](const float* w, int O, int I, int KH, int KW, int Npad, int Kpad) {
    _Float16* out = allocH((size_t)Npad * Kpad);
    k_wconv<<<g256((long long)Npad * Kpad), 256, 0, stream>>>(w, out, O, I, KH, KW, Npad, Kpad);
    return out;
  };
  auto gemm_full = [&](const _Float16* A, const _Float16* Bw, float* out,
                       _Float16* out2, int ld2, int tr2, const float* bias,
                       const float* resid, int M, int Nout, int Npad, int Kpad,
                       int act, float scale) {
    dim3 grid((unsigned)(M / 128), (unsigned)(Npad / 64));
    k_gemm<<<grid, 128, 0, stream>>>(A, Bw, out, out2, ld2, tr2, bias, resid,
                                     M, Nout, Kpad, act, scale);
  };
  auto gemm = [&](const _Float16* A, const _Float16* Bw, float* out,
                  _Float16* out2, int ld2, const float* bias, const float* resid,
                  int M, int Nout, int Npad, int Kpad, int act, float scale) {
    gemm_full(A, Bw, out, out2, ld2, 0, bias, resid, M, Nout, Npad, Kpad, act, scale);
  };
  auto im2col = [&](const float* p0, int C0, const float* p1, int C1,
                    const float* p2, int C2, _Float16* out, int Kpad,
                    int KH, int KW, int padH, int padW) {
    k_im2col3<<<g256((long long)HWPIX * Kpad), 256, 0, stream>>>(
        p0, C0, p1, C1, p2, C2, out, Kpad, KH, KW, padH, padW);
  };
  auto pack3 = [&](const float* p0, int C0, const float* p1, int C1,
                   const float* p2, int C2, float* out, _Float16* outh, int Cpad) {
    k_pack3<<<g256((long long)HWPIX * Cpad), 256, 0, stream>>>(p0, C0, p1, C1, p2, C2, out, outh, Cpad);
  };

  // ---------------- weights -> padded f16 ----------------
  _Float16* w_proj = wconv(P(P_PROJ_W), 256, 256, 1, 1, 256, 256);
  _Float16* w_fte1 = wconv(P(P_FTE1_W), 64, 81, 1, 1, 64, 96);
  _Float16* w_fte2 = wconv(P(P_FTE2_W), 64, 64, 1, 1, 64, 64);
  _Float16* w_caq  = wconv(P(P_CA_Q_W), 64, 64, 1, 1, 64, 64);
  _Float16* w_cak  = wconv(P(P_CA_K_W), 64, 128, 1, 1, 64, 128);
  _Float16* w_cav  = wconv(P(P_CA_V_W), 64, 128, 1, 1, 64, 128);
  _Float16* w_cap  = wconv(P(P_CA_P_W), 64, 128, 1, 1, 64, 128);
  _Float16* w_caf1 = wconv(P(P_CA_F1_W), 64, 64, 1, 1, 64, 64);
  _Float16* w_caf2 = wconv(P(P_CA_F2_W), 64, 64, 1, 1, 64, 64);
  _Float16* w_qk   = wconv(P(P_ATT_QK_W), 256, 128, 1, 1, 256, 128);
  _Float16* w_ec1  = wconv(P(P_ENC_C1_W), 256, 145, 1, 1, 256, 160);
  _Float16* w_ec2  = wconv(P(P_ENC_C2_W), 192, 256, 3, 3, 192, 2304);
  _Float16* w_ef1  = wconv(P(P_ENC_F1_W), 128, 2, 7, 7, 128, 128);
  _Float16* w_ef2  = wconv(P(P_ENC_F2_W), 64, 128, 3, 3, 64, 1152);
  _Float16* w_eo   = wconv(P(P_ENC_O_W), 126, 256, 3, 3, 128, 2304);
  _Float16* w_av   = wconv(P(P_AGG_V_W), 128, 128, 1, 1, 128, 128);
  _Float16* w_gz1  = wconv(P(P_GRU_Z1_W), 128, 512, 1, 5, 128, 2560);
  _Float16* w_gr1  = wconv(P(P_GRU_R1_W), 128, 512, 1, 5, 128, 2560);
  _Float16* w_gq1  = wconv(P(P_GRU_Q1_W), 128, 512, 1, 5, 128, 2560);
  _Float16* w_gz2  = wconv(P(P_GRU_Z2_W), 128, 512, 5, 1, 128, 2560);
  _Float16* w_gr2  = wconv(P(P_GRU_R2_W), 128, 512, 5, 1, 128, 2560);
  _Float16* w_gq2  = wconv(P(P_GRU_Q2_W), 128, 512, 5, 1, 128, 2560);
  _Float16* w_fh1  = wconv(P(P_FH1_W), 256, 128, 3, 3, 256, 1152);
  _Float16* w_fh2  = wconv(P(P_FH2_W), 2, 256, 3, 3, 64, 2304);
  _Float16* w_m1   = wconv(P(P_MASK1_W), 256, 128, 3, 3, 256, 1152);
  _Float16* w_m2   = wconv(P(P_MASK2_W), 576, 256, 1, 1, 576, 256);
  _Float16* mem_h  = wconv(P(IN_COST_MEMORY), 49152, 128, 1, 1, 49152, 128);

  // ---------------- activations / scratch ----------------
  _Float16* ctx_h  = allocH((size_t)HWPIX * 256);
  float*    proj   = allocF((size_t)HWPIX * 256);
  float*    net    = allocF((size_t)HWPIX * 128);
  float*    inp    = allocF((size_t)HWPIX * 128);
  _Float16* inp_h  = allocH((size_t)HWPIX * 128);
  float*    qkbuf  = allocF((size_t)HWPIX * 256);
  _Float16* qh     = allocH((size_t)HWPIX * 128);
  _Float16* kh     = allocH((size_t)HWPIX * 128);
  float*    kmem   = allocF((size_t)49152 * 64);
  float*    vmem   = allocF((size_t)49152 * 64);
  float*    coords = allocF((size_t)HWPIX * 2);
  float*    flow   = allocF((size_t)HWPIX * 2);
  float*    cfwd   = allocF((size_t)HWPIX * 81);
  _Float16* cfwd_h = allocH((size_t)HWPIX * 96);
  float*    q1f    = allocF((size_t)HWPIX * 64);
  _Float16* q1_h   = allocH((size_t)HWPIX * 64);
  float*    query  = allocF((size_t)HWPIX * 64);
  _Float16* qln_h  = allocH((size_t)HWPIX * 64);
  float*    qproj  = allocF((size_t)HWPIX * 64);
  float*    xo     = allocF((size_t)HWPIX * 64);
  _Float16* xoq_h  = allocH((size_t)HWPIX * 128);
  float*    x1     = allocF((size_t)HWPIX * 64);
  _Float16* xln_h  = allocH((size_t)HWPIX * 64);
  float*    h1f    = allocF((size_t)HWPIX * 64);
  _Float16* h1_h   = allocH((size_t)HWPIX * 64);
  float*    cg     = allocF((size_t)HWPIX * 64);
  _Float16* corr_h = allocH((size_t)HWPIX * 160);
  float*    c1     = allocF((size_t)HWPIX * 256);
  float*    c2     = allocF((size_t)HWPIX * 192);
  float*    f1o    = allocF((size_t)HWPIX * 128);
  float*    f2o    = allocF((size_t)HWPIX * 64);
  float*    oo     = allocF((size_t)HWPIX * 126);
  float*    mf     = allocF((size_t)HWPIX * 128);
  _Float16* mf_h   = allocH((size_t)HWPIX * 128);
  float*    vf     = allocF((size_t)HWPIX * 128);
  _Float16* vbt    = allocH((size_t)HWPIX * 128);  // V transposed [128, HW]
  float*    agg    = allocF((size_t)HWPIX * 128);
  float*    mfg    = allocF((size_t)HWPIX * 128);
  float*    xcat   = allocF((size_t)HWPIX * 384);
  float*    zbuf   = allocF((size_t)HWPIX * 128);
  float*    rbuf   = allocF((size_t)HWPIX * 128);
  float*    rh     = allocF((size_t)HWPIX * 128);
  float*    qbuf   = allocF((size_t)HWPIX * 128);
  float*    fh     = allocF((size_t)HWPIX * 256);
  float*    msk1   = allocF((size_t)HWPIX * 256);
  _Float16* msk1_h = allocH((size_t)HWPIX * 256);
  float*    upmask = allocF((size_t)HWPIX * 576);
  float*    dflow  = allocF((size_t)HWPIX * 2);
  _Float16* col    = allocH((size_t)HWPIX * 2560);

  // ---------------- prologue (iteration-invariant) ----------------
  k_nchw2hwc<<<g256((long long)HWPIX * 256), 256, 0, stream>>>(P(IN_CONTEXT), ctx_h, 256);
  gemm(ctx_h, w_proj, proj, nullptr, 0, P(P_PROJ_B), nullptr, HWPIX, 256, 256, 256, 0, 1.0f);
  k_split_ctx<<<g256((long long)HWPIX * 128), 256, 0, stream>>>(proj, net, inp, inp_h);
  gemm(inp_h, w_qk, qkbuf, nullptr, 0, nullptr, nullptr, HWPIX, 256, 256, 128, 0, 1.0f);
  k_split_qk<<<g256((long long)HWPIX * 128), 256, 0, stream>>>(qkbuf, qh, kh);
  gemm(mem_h, w_cak, kmem, nullptr, 0, P(P_CA_K_B), nullptr, 49152, 64, 64, 128, 0, 1.0f);
  gemm(mem_h, w_cav, vmem, nullptr, 0, P(P_CA_V_B), nullptr, 49152, 64, 64, 128, 0, 1.0f);
  k_cinit<<<g256(HWPIX), 256, 0, stream>>>(coords);

  const float* cmaps = P(IN_COST_MAPS);

  for (int it = 0; it < 2; ++it) {
    k_flowk<<<g256(HWPIX), 256, 0, stream>>>(coords, flow);
    // cost volume sampling + feature token encoding
    k_sample<<<g256((long long)HWPIX * 96), 256, 0, stream>>>(cmaps, coords, cfwd, cfwd_h);
    gemm(cfwd_h, w_fte1, q1f, q1_h, 64, P(P_FTE1_B), nullptr, HWPIX, 64, 64, 96, 2, 1.0f);
    gemm(q1_h, w_fte2, query, nullptr, 0, P(P_FTE2_B), nullptr, HWPIX, 64, 64, 64, 0, 1.0f);
    // cross attention with cost memory
    k_qln_pos<<<g256(HWPIX), 256, 0, stream>>>(query, coords, P(P_CA_N1_G), P(P_CA_N1_B), qln_h);
    gemm(qln_h, w_caq, qproj, nullptr, 0, P(P_CA_Q_B), nullptr, HWPIX, 64, 64, 64, 0, 1.0f);
    k_sattn<<<g256((long long)HWPIX * 8), 256, 0, stream>>>(qproj, kmem, vmem, xo);
    pack3(xo, 64, query, 64, nullptr, 0, nullptr, xoq_h, 128);
    gemm(xoq_h, w_cap, x1, nullptr, 0, P(P_CA_P_B), query, HWPIX, 64, 64, 128, 0, 1.0f);
    k_ln64<<<g256(HWPIX), 256, 0, stream>>>(x1, P(P_CA_N2_G), P(P_CA_N2_B), xln_h);
    gemm(xln_h, w_caf1, h1f, h1_h, 64, P(P_CA_F1_B), nullptr, HWPIX, 64, 64, 64, 2, 1.0f);
    gemm(h1_h, w_caf2, cg, nullptr, 0, P(P_CA_F2_B), x1, HWPIX, 64, 64, 64, 0, 1.0f);
    // motion encoder
    im2col(cg, 64, cfwd, 81, nullptr, 0, corr_h, 160, 1, 1, 0, 0);
    gemm(corr_h, w_ec1, c1, nullptr, 0, P(P_ENC_C1_B), nullptr, HWPIX, 256, 256, 160, 1, 1.0f);
    im2col(c1, 256, nullptr, 0, nullptr, 0, col, 2304, 3, 3, 1, 1);
    gemm(col, w_ec2, c2, nullptr, 0, P(P_ENC_C2_B), nullptr, HWPIX, 192, 192, 2304, 1, 1.0f);
    im2col(flow, 2, nullptr, 0, nullptr, 0, col, 128, 7, 7, 3, 3);
    gemm(col, w_ef1, f1o, nullptr, 0, P(P_ENC_F1_B), nullptr, HWPIX, 128, 128, 128, 1, 1.0f);
    im2col(f1o, 128, nullptr, 0, nullptr, 0, col, 1152, 3, 3, 1, 1);
    gemm(col, w_ef2, f2o, nullptr, 0, P(P_ENC_F2_B), nullptr, HWPIX, 64, 64, 1152, 1, 1.0f);
    im2col(c2, 192, f2o, 64, nullptr, 0, col, 2304, 3, 3, 1, 1);
    gemm(col, w_eo, oo, nullptr, 0, P(P_ENC_O_B), nullptr, HWPIX, 126, 128, 2304, 1, 1.0f);
    pack3(oo, 126, flow, 2, nullptr, 0, mf, mf_h, 128);
    // GMA aggregation (fused flash attention, V written transposed)
    gemm_full(mf_h, w_av, vf, vbt, HWPIX, 1, nullptr, nullptr, HWPIX, 128, 128, 128, 0, 1.0f);
    k_flash<<<96, 128, 0, stream>>>(qh, kh, vbt, agg, HWPIX);
    k_mfg<<<g256((long long)HWPIX * 128), 256, 0, stream>>>(mf, agg, P(P_AGG_GAMMA), mfg);
    pack3(inp, 128, mf, 128, mfg, 128, xcat, nullptr, 384);
    // SepConv GRU, horizontal (1x5)
    im2col(net, 128, xcat, 384, nullptr, 0, col, 2560, 1, 5, 0, 2);
    gemm(col, w_gz1, zbuf, nullptr, 0, P(P_GRU_Z1_B), nullptr, HWPIX, 128, 128, 2560, 4, 1.0f);
    gemm(col, w_gr1, rbuf, nullptr, 0, P(P_GRU_R1_B), nullptr, HWPIX, 128, 128, 2560, 4, 1.0f);
    k_rh<<<g256((long long)HWPIX * 128), 256, 0, stream>>>(rbuf, net, rh);
    im2col(rh, 128, xcat, 384, nullptr, 0, col, 2560, 1, 5, 0, 2);
    gemm(col, w_gq1, qbuf, nullptr, 0, P(P_GRU_Q1_B), nullptr, HWPIX, 128, 128, 2560, 3, 1.0f);
    k_gruout<<<g256((long long)HWPIX * 128), 256, 0, stream>>>(net, zbuf, qbuf);
    // SepConv GRU, vertical (5x1)
    im2col(net, 128, xcat, 384, nullptr, 0, col, 2560, 5, 1, 2, 0);
    gemm(col, w_gz2, zbuf, nullptr, 0, P(P_GRU_Z2_B), nullptr, HWPIX, 128, 128, 2560, 4, 1.0f);
    gemm(col, w_gr2, rbuf, nullptr, 0, P(P_GRU_R2_B), nullptr, HWPIX, 128, 128, 2560, 4, 1.0f);
    k_rh<<<g256((long long)HWPIX * 128), 256, 0, stream>>>(rbuf, net, rh);
    im2col(rh, 128, xcat, 384, nullptr, 0, col, 2560, 5, 1, 2, 0);
    gemm(col, w_gq2, qbuf, nullptr, 0, P(P_GRU_Q2_B), nullptr, HWPIX, 128, 128, 2560, 3, 1.0f);
    k_gruout<<<g256((long long)HWPIX * 128), 256, 0, stream>>>(net, zbuf, qbuf);
    // flow head + mask head
    im2col(net, 128, nullptr, 0, nullptr, 0, col, 1152, 3, 3, 1, 1);
    gemm(col, w_fh1, fh, nullptr, 0, P(P_FH1_B), nullptr, HWPIX, 256, 256, 1152, 1, 1.0f);
    gemm(col, w_m1, msk1, msk1_h, 256, P(P_MASK1_B), nullptr, HWPIX, 256, 256, 1152, 1, 1.0f);
    im2col(fh, 256, nullptr, 0, nullptr, 0, col, 2304, 3, 3, 1, 1);
    gemm(col, w_fh2, dflow, nullptr, 0, P(P_FH2_B), nullptr, HWPIX, 2, 64, 2304, 0, 1.0f);
    gemm(msk1_h, w_m2, upmask, nullptr, 0, P(P_MASK2_B), nullptr, HWPIX, 576, 576, 256, 0, 0.25f);
    k_cadd<<<g256((long long)HWPIX * 2), 256, 0, stream>>>(coords, dflow);
  }

  // outputs: flow_up (1,2,512,768) then flow (1,2,64,96)
  float* outp = (float*)d_out;
  k_flowk<<<g256(HWPIX), 256, 0, stream>>>(coords, flow);
  k_upsample<<<g256((long long)HWPIX * 64), 256, 0, stream>>>(flow, upmask, outp);
  k_outflow<<<g256((long long)2 * HWPIX), 256, 0, stream>>>(flow, outp + 786432);
}
